// GINEncoder_14439680049632
// MI455X (gfx1250) — matrix operand order repaired and checked by reading, not compiled, not run on hardware
//
#include <hip/hip_runtime.h>
#include <hip/hip_bf16.h>

// ---------------------------------------------------------------------------
// GIN encoder for MI455X (gfx1250, wave32, WMMA).
//  - MLP GEMMs run on v_wmma_f32_16x16x32_bf16 (fp32 accumulate).
//  - h (51.2MB) is L2-resident (192MB L2); scatter-add is L2-atomic bound.
//  - BN folded to a*x+c; weights pre-transposed to bf16 [n][k] so WMMA A/B
//    fragments are contiguous 32-bit LDS loads per the CDNA5 VGPR layouts.
// ---------------------------------------------------------------------------

typedef __attribute__((ext_vector_type(16))) __bf16 v16bf;
typedef __attribute__((ext_vector_type(8)))  float  v8f;

constexpr int   Nn   = 100000;
constexpr int   Ee   = 3200000;
constexpr int   Gg   = 512;
constexpr int   Ll   = 4;
constexpr int   Hh   = 128;
constexpr int   H2   = 256;
constexpr int   FIN  = 9;
constexpr int   BM   = 64;      // node rows per block in MLP kernel
constexpr int   MLP_THREADS = 256;
constexpr size_t SMEM_BYTES = (size_t)(BM*Hh + BM*H2 + Hh*H2) * 2; // 112KB

// ---------------- fragment loaders (bf16, row-major [row][k] in LDS) -------
union Frag { v16bf v; unsigned int d[8]; };

// A fragment: 16x32 bf16. lane m=lane&15 holds row m; half=lane>>4.
//   d[0..3] -> K = kb + 8*half + {0..7};  d[4..7] -> K = kb + 8*half + 16 + {0..7}
__device__ inline v16bf load_a_frag(const __bf16* row, int kb, int half) {
  Frag f;
  const __bf16* p = row + kb + 8*half;
#pragma unroll
  for (int i = 0; i < 4; ++i) f.d[i]     = *(const unsigned int*)(p + 2*i);
#pragma unroll
  for (int i = 0; i < 4; ++i) f.d[4 + i] = *(const unsigned int*)(p + 16 + 2*i);
  return f.v;
}

// B fragment: 32x16 bf16, weights stored transposed [n][k]. lane n=lane&15.
//   d[0..3] -> K = kb + 8*half + {0..7};  d[4..7] -> K = kb + 8*half + 16 + {0..7}
__device__ inline v16bf load_b_frag(const __bf16* row, int kb, int half) {
  Frag f;
  const __bf16* p = row + kb + 8*half;
#pragma unroll
  for (int i = 0; i < 4; ++i) f.d[i]     = *(const unsigned int*)(p + 2*i);
#pragma unroll
  for (int i = 0; i < 4; ++i) f.d[4 + i] = *(const unsigned int*)(p + 16 + 2*i);
  return f.v;
}

// ---------------- prep kernels ---------------------------------------------
__global__ void wt_prep(const float* __restrict__ W1, const float* __restrict__ W2,
                        __bf16* __restrict__ W1t, __bf16* __restrict__ W2t) {
  int i = blockIdx.x * blockDim.x + threadIdx.x;
  const int per = Ll * Hh * H2;
  if (i < per) {
    int l = i / (Hh * H2), r = i % (Hh * H2);
    int n = r / Hh, k = r % Hh;                    // W1t[l][n][k] = W1[l][k][n]
    W1t[i] = (__bf16)W1[(size_t)l * Hh * H2 + (size_t)k * H2 + n];
  } else if (i < 2 * per) {
    int j = i - per;
    int l = j / (Hh * H2), r = j % (Hh * H2);
    int n = r / H2, k = r % H2;                    // W2t[l][n][k] = W2[l][k][n]
    W2t[j] = (__bf16)W2[(size_t)l * Hh * H2 + (size_t)k * Hh + n];
  }
}

__global__ void bn_prep(const float* __restrict__ b1, const float* __restrict__ g1,
                        const float* __restrict__ be1, const float* __restrict__ m1,
                        const float* __restrict__ v1,
                        const float* __restrict__ b2, const float* __restrict__ g2,
                        const float* __restrict__ be2, const float* __restrict__ m2,
                        const float* __restrict__ v2,
                        float* __restrict__ a1, float* __restrict__ c1,
                        float* __restrict__ a2, float* __restrict__ c2) {
  int i = blockIdx.x * blockDim.x + threadIdx.x;
  if (i < Ll * H2) {
    float a = g1[i] * rsqrtf(v1[i] + 1e-5f);
    a1[i] = a;  c1[i] = a * (b1[i] - m1[i]) + be1[i];
  } else if (i < Ll * H2 + Ll * Hh) {
    int j = i - Ll * H2;
    float a = g2[j] * rsqrtf(v2[j] + 1e-5f);
    a2[j] = a;  c2[j] = a * (b2[j] - m2[j]) + be2[j];
  }
}

// ---------------- embedding: h = x @ Wemb + bemb ---------------------------
__global__ void embed(const float* __restrict__ x, const float* __restrict__ Wemb,
                      const float* __restrict__ bemb, float* __restrict__ h) {
  int i = blockIdx.x * blockDim.x + threadIdx.x;
  if (i >= Nn * Hh) return;
  int n = i >> 7, c = i & (Hh - 1);
  const float* xr = x + (size_t)n * FIN;
  float acc = bemb[c];
#pragma unroll
  for (int k = 0; k < FIN; ++k) acc = fmaf(xr[k], Wemb[k * Hh + c], acc);
  h[i] = acc;
}

// ---------------- z = (1+eps[l]) * h ---------------------------------------
__global__ void scale_init(const float* __restrict__ h, float* __restrict__ z,
                           const float* __restrict__ eps, int layer) {
  int i = blockIdx.x * blockDim.x + threadIdx.x;
  if (i < Nn * Hh) z[i] = (1.0f + eps[layer]) * h[i];
}

// ---------------- edge scatter: z[dst] += h[src] (32 lanes/edge, float4) ---
__global__ void scatter_add(const float* __restrict__ h, const long long* __restrict__ src,
                            const long long* __restrict__ dst, float* __restrict__ z) {
  int t = blockIdx.x * blockDim.x + threadIdx.x;
  int e = t >> 5, lane = t & 31;
  if (e >= Ee) return;
  long long s = src[e], d = dst[e];
  float4 v = ((const float4*)(h + (size_t)s * Hh))[lane];
  float* p = z + (size_t)d * Hh + lane * 4;
  atomicAdd(p + 0, v.x); atomicAdd(p + 1, v.y);
  atomicAdd(p + 2, v.z); atomicAdd(p + 3, v.w);
}

// ---------------- fused GIN MLP: h = relu(bn2(relu(bn1(z@W1))@W2)) ---------
__global__ __launch_bounds__(MLP_THREADS)
void gin_mlp(const float* __restrict__ z, float* __restrict__ hout,
             const __bf16* __restrict__ W1t,   // [H2][H]  (transposed, bf16)
             const __bf16* __restrict__ W2t,   // [H][H2]  (transposed, bf16)
             const float* __restrict__ a1, const float* __restrict__ c1,   // [H2]
             const float* __restrict__ a2, const float* __restrict__ c2) { // [H]
  extern __shared__ char smem[];
  __bf16* sA = (__bf16*)smem;                                  // BM x H   16KB
  __bf16* sZ = (__bf16*)(smem + (size_t)BM * Hh * 2);          // BM x H2  32KB
  __bf16* sW = (__bf16*)(smem + (size_t)(BM * Hh + BM * H2) * 2); // 64KB

  const int tid  = threadIdx.x;
  const int wave = tid >> 5;
  const int lane = tid & 31;
  const int half = lane >> 4;
  const int l16  = lane & 15;
  const int row0 = blockIdx.x * BM;

  // stage in z tile (fp32 -> bf16) and W1t
  for (int i = tid; i < BM * Hh / 4; i += MLP_THREADS) {
    int r = (i * 4) / Hh, c = (i * 4) % Hh;
    float4 v = make_float4(0.f, 0.f, 0.f, 0.f);
    if (row0 + r < Nn) v = *(const float4*)(z + (size_t)(row0 + r) * Hh + c);
    __bf16* p = sA + r * Hh + c;
    p[0] = (__bf16)v.x; p[1] = (__bf16)v.y; p[2] = (__bf16)v.z; p[3] = (__bf16)v.w;
  }
  for (int i = tid; i < Hh * H2 * 2 / 16; i += MLP_THREADS)
    ((uint4*)sW)[i] = ((const uint4*)W1t)[i];
  __syncthreads();

  // ---- GEMM1: [BM,128] x [128,256] -> BN1 -> ReLU -> sZ (bf16) ----
  // 64 output tiles of 16x16; 8 waves x 8 tiles; K-loop = 4 WMMAs each.
  for (int t = 0; t < 8; ++t) {
    int tile = wave * 8 + t;
    int tm = tile & 3, tn = tile >> 2;
    v8f acc = {};
    const __bf16* arow = sA + (size_t)(tm * 16 + l16) * Hh;
    const __bf16* brow = sW + (size_t)(tn * 16 + l16) * Hh;
#pragma unroll
    for (int kb = 0; kb < Hh; kb += 32) {
      v16bf af = load_a_frag(arow, kb, half);
      v16bf bf = load_b_frag(brow, kb, half);
      acc = __builtin_amdgcn_wmma_f32_16x16x32_bf16(false, af, false, bf,
                                                    (short)0, acc, false, false);
    }
    int col = tn * 16 + l16;
    float av = a1[col], cv = c1[col];
#pragma unroll
    for (int r = 0; r < 8; ++r) {
      int rr = tm * 16 + r + 8 * half;
      float y = fmaf(av, acc[r], cv);
      sZ[rr * H2 + col] = (__bf16)(y > 0.f ? y : 0.f);
    }
  }
  __syncthreads();

  // swap in W2t (reuse sW region)
  for (int i = tid; i < Hh * H2 * 2 / 16; i += MLP_THREADS)
    ((uint4*)sW)[i] = ((const uint4*)W2t)[i];
  __syncthreads();

  // ---- GEMM2: [BM,256] x [256,128] -> BN2 -> ReLU -> hout (fp32) ----
  // 32 output tiles; 8 waves x 4 tiles; K-loop = 8 WMMAs each.
  for (int t = 0; t < 4; ++t) {
    int tm = t, tn = wave;
    v8f acc = {};
    const __bf16* arow = sZ + (size_t)(tm * 16 + l16) * H2;
    const __bf16* brow = sW + (size_t)(tn * 16 + l16) * H2;
#pragma unroll
    for (int kb = 0; kb < H2; kb += 32) {
      v16bf af = load_a_frag(arow, kb, half);
      v16bf bf = load_b_frag(brow, kb, half);
      acc = __builtin_amdgcn_wmma_f32_16x16x32_bf16(false, af, false, bf,
                                                    (short)0, acc, false, false);
    }
    int col = tn * 16 + l16;
    float av = a2[col], cv = c2[col];
#pragma unroll
    for (int r = 0; r < 8; ++r) {
      int rr = tm * 16 + r + 8 * half;
      float y = fmaf(av, acc[r], cv);
      y = y > 0.f ? y : 0.f;
      if (row0 + rr < Nn) hout[(size_t)(row0 + rr) * Hh + col] = y;
    }
  }
}

// ---------------- mean pool ------------------------------------------------
__global__ void pool_zero(float* __restrict__ sums, float* __restrict__ counts) {
  int i = blockIdx.x * blockDim.x + threadIdx.x;
  if (i < Gg * Hh) sums[i] = 0.f;
  if (i < Gg) counts[i] = 0.f;
}

__global__ void pool_acc(const float* __restrict__ h, const long long* __restrict__ batch,
                         float* __restrict__ sums, float* __restrict__ counts) {
  int t = blockIdx.x * blockDim.x + threadIdx.x;
  int n = t >> 5, lane = t & 31;
  if (n >= Nn) return;
  int g = (int)batch[n];
  float4 v = ((const float4*)(h + (size_t)n * Hh))[lane];
  float* s = sums + (size_t)g * Hh + lane * 4;
  atomicAdd(s + 0, v.x); atomicAdd(s + 1, v.y);
  atomicAdd(s + 2, v.z); atomicAdd(s + 3, v.w);
  if (lane == 0) atomicAdd(counts + g, 1.0f);
}

__global__ void pool_div(const float* __restrict__ sums, const float* __restrict__ counts,
                         float* __restrict__ out) {
  int i = blockIdx.x * blockDim.x + threadIdx.x;
  if (i < Gg * Hh) out[i] = sums[i] / fmaxf(counts[i / Hh], 1.0f);
}

// ---------------------------------------------------------------------------
extern "C" void kernel_launch(void* const* d_in, const int* in_sizes, int n_in,
                              void* d_out, int out_size, void* d_ws, size_t ws_size,
                              hipStream_t stream) {
  const float*     x     = (const float*)d_in[0];
  const long long* ei    = (const long long*)d_in[1];   // [2,E]
  const long long* batch = (const long long*)d_in[2];
  const float*     Wemb  = (const float*)d_in[3];
  const float*     bemb  = (const float*)d_in[4];
  const float*     eps   = (const float*)d_in[5];
  const float*     W1    = (const float*)d_in[6];
  const float*     b1    = (const float*)d_in[7];
  const float*     g1    = (const float*)d_in[8];
  const float*     be1   = (const float*)d_in[9];
  const float*     m1    = (const float*)d_in[10];
  const float*     v1    = (const float*)d_in[11];
  const float*     W2    = (const float*)d_in[12];
  const float*     b2    = (const float*)d_in[13];
  const float*     g2    = (const float*)d_in[14];
  const float*     be2   = (const float*)d_in[15];
  const float*     m2    = (const float*)d_in[16];
  const float*     v2    = (const float*)d_in[17];
  float* out = (float*)d_out;

  // workspace carve-up (256B aligned)
  char* p = (char*)d_ws;
  auto carve = [&](size_t bytes) { char* r = p; p += (bytes + 255) & ~(size_t)255; return r; };
  float*  h      = (float*)carve((size_t)Nn * Hh * 4);
  float*  zbuf   = (float*)carve((size_t)Nn * Hh * 4);
  __bf16* W1t    = (__bf16*)carve((size_t)Ll * Hh * H2 * 2);
  __bf16* W2t    = (__bf16*)carve((size_t)Ll * Hh * H2 * 2);
  float*  a1     = (float*)carve((size_t)Ll * H2 * 4);
  float*  c1     = (float*)carve((size_t)Ll * H2 * 4);
  float*  a2     = (float*)carve((size_t)Ll * Hh * 4);
  float*  c2     = (float*)carve((size_t)Ll * Hh * 4);
  float*  sums   = (float*)carve((size_t)Gg * Hh * 4);
  float*  counts = (float*)carve((size_t)Gg * 4);

  (void)hipFuncSetAttribute(reinterpret_cast<const void*>(&gin_mlp),
                            hipFuncAttributeMaxDynamicSharedMemorySize,
                            (int)SMEM_BYTES);

  // parameter prep (cheap, graph-replayed each call: deterministic)
  {
    int n = 2 * Ll * Hh * H2;
    wt_prep<<<(n + 255) / 256, 256, 0, stream>>>(W1, W2, W1t, W2t);
    int m = Ll * H2 + Ll * Hh;
    bn_prep<<<(m + 255) / 256, 256, 0, stream>>>(b1, g1, be1, m1, v1,
                                                 b2, g2, be2, m2, v2,
                                                 a1, c1, a2, c2);
  }

  embed<<<(Nn * Hh + 255) / 256, 256, 0, stream>>>(x, Wemb, bemb, h);

  for (int l = 0; l < Ll; ++l) {
    scale_init<<<(Nn * Hh + 255) / 256, 256, 0, stream>>>(h, zbuf, eps, l);
    scatter_add<<<((size_t)Ee * 32 + 255) / 256, 256, 0, stream>>>(h, ei, ei + Ee, zbuf);
    gin_mlp<<<(Nn + BM - 1) / BM, MLP_THREADS, SMEM_BYTES, stream>>>(
        zbuf, h,
        W1t + (size_t)l * Hh * H2, W2t + (size_t)l * Hh * H2,
        a1 + (size_t)l * H2, c1 + (size_t)l * H2,
        a2 + (size_t)l * Hh, c2 + (size_t)l * Hh);
  }

  pool_zero<<<(Gg * Hh + 255) / 256, 256, 0, stream>>>(sums, counts);
  pool_acc<<<((size_t)Nn * 32 + 255) / 256, 256, 0, stream>>>(h, batch, sums, counts);
  pool_div<<<(Gg * Hh + 255) / 256, 256, 0, stream>>>(sums, counts, out);
}
